// LengthRegulator_47210280517762
// MI455X (gfx1250) — compile-verified, matrix operand order checked by reading, not run on hardware
//
#include <hip/hip_runtime.h>
#include <hip/hip_bf16.h>
#include <math.h>

// ---------------- problem constants (static in the reference) ----------------
#define B_   4
#define T_   2048
#define D_   384          // input feature dim (K of GEMM1)
#define H_   192          // hidden dim       (N of GEMM1)
#define NKT  (D_ / 32)    // 12 WMMA K-steps of 32
#define NNT  (H_ / 16)    // 12 N-tiles of 16
#define TOTAL_LEN (B_ * T_ * 8)   // 65536, == reference total_length

typedef __attribute__((ext_vector_type(16))) _Float16 v16h;
typedef __attribute__((ext_vector_type(8)))  float    v8f;
typedef __attribute__((ext_vector_type(4)))  float    f4;

// ---------------------------------------------------------------------------
// Kernel 1: fused duration MLP via v_wmma_f32_16x16x32_f16.
// One wave (32 threads) computes 16 tokens. A fragments hold x rows in f16,
// B fragments hold w1 columns in f16; after each 16x16 C tile we fuse
// bias + ReLU + (dot with w2 column scale) into per-lane partial sums.
// ---------------------------------------------------------------------------
__global__ __launch_bounds__(32)
void dur_mlp_wmma(const float* __restrict__ x,  const float* __restrict__ w1,
                  const float* __restrict__ b1, const float* __restrict__ w2,
                  const float* __restrict__ b2, float* __restrict__ durOut,
                  int* __restrict__ duri) {
  const int tile    = blockIdx.x;      // 0 .. (B*T/16 - 1)
  const int rowBase = tile * 16;       // flattened (b,t) token index
  const int lane    = threadIdx.x;     // wave32
  const int m       = lane & 15;       // row (A) / col (B) within tile
  const int halfSel = lane >> 4;       // K-half selector per ISA layout

  // Preload all A fragments for this 16-token tile (K = 384 -> 12 frags).
  // 16-bit A 16x32 layout: lanes 0-15 hold K {0..7,16..23}, lanes 16-31
  // hold K {8..15,24..31} for row M = lane%16.
  v16h afr[NKT];
#pragma unroll
  for (int kt = 0; kt < NKT; ++kt) {
    const float* xr = x + (size_t)(rowBase + m) * D_ + kt * 32;
#pragma unroll
    for (int e = 0; e < 16; ++e) {
      const int k = (e < 8 ? e : e + 8) + halfSel * 8;
      afr[kt][e] = (_Float16)xr[k];
    }
  }

  float partial[8];
#pragma unroll
  for (int r = 0; r < 8; ++r) partial[r] = 0.0f;

  for (int nt = 0; nt < NNT; ++nt) {
    const int n0 = nt * 16;
    v8f c = {};
#pragma unroll
    for (int kt = 0; kt < NKT; ++kt) {
      // B fragment: column n = n0 + (lane%16), K mapping mirrors A.
      v16h bfr;
      const float* wc = w1 + (size_t)(kt * 32) * H_ + n0 + m;
#pragma unroll
      for (int e = 0; e < 16; ++e) {
        const int k = (e < 8 ? e : e + 8) + halfSel * 8;
        bfr[e] = (_Float16)wc[(size_t)k * H_];
      }
      c = __builtin_amdgcn_wmma_f32_16x16x32_f16(
          /*neg_a=*/false, afr[kt], /*neg_b=*/false, bfr,
          /*c_mod=*/(short)0, c, /*reuse_a=*/false, /*reuse_b=*/false);
    }
    // C layout: VGPR r, lane l -> row = r + 8*(l/16), col = l%16.
    const float wscale = w2[n0 + m];   // second-layer weight for this column
    const float bias1  = b1[n0 + m];
#pragma unroll
    for (int r = 0; r < 8; ++r) {
      float v = c[r] + bias1;
      v = v > 0.0f ? v : 0.0f;         // ReLU
      partial[r] += v * wscale;        // fold GEMM2 (192 -> 1)
    }
  }

  // Reduce the 16 columns held by each half-wave (masks < 16 stay in-half).
#pragma unroll
  for (int r = 0; r < 8; ++r) {
    float v = partial[r];
    v += __shfl_xor(v, 1, 32);
    v += __shfl_xor(v, 2, 32);
    v += __shfl_xor(v, 4, 32);
    v += __shfl_xor(v, 8, 32);
    partial[r] = v;
  }

  if ((lane & 15) == 0) {              // lane 0 -> rows 0..7, lane 16 -> 8..15
    const float bb = b2[0];
#pragma unroll
    for (int r = 0; r < 8; ++r) {
      const int row = rowBase + halfSel * 8 + r;   // flattened (b,t)
      const float z  = partial[r] + bb;
      const float sp = z > 20.0f ? z : log1pf(__expf(z));   // softplus
      const float dur = rintf(fmaxf(sp, 8.0f));             // clamp + RNE round
      durOut[row] = dur;
      if (row < T_) duri[row] = (int)dur;          // batch-0 durations for scan
    }
  }
}

// ---------------------------------------------------------------------------
// Kernel 2: serial inclusive cumsum of the 2048 batch-0 durations (tiny).
// ---------------------------------------------------------------------------
__global__ void dur_scan(const int* __restrict__ duri, int* __restrict__ cum) {
  if (threadIdx.x == 0) {
    int acc = 0;
    for (int t = 0; t < T_; ++t) { acc += duri[t]; cum[t] = acc; }
  }
}

// ---------------------------------------------------------------------------
// Kernel 3: length-regulator expand. One 96-thread block per output row
// (D = 384 floats = 96 float4). Thread 0 binary-searches cum (8 KB, cache
// resident), broadcasts; all threads stream the row out with non-temporal
// stores (402 MB of pure streaming writes -> keep L2 for x).
// ---------------------------------------------------------------------------
__global__ __launch_bounds__(96)
void expand_k(const float* __restrict__ x, const int* __restrict__ cum,
              float* __restrict__ out) {
  __shared__ int sIdx;
  const int p = (int)blockIdx.x;       // output position 0..TOTAL_LEN-1
  const int b = (int)blockIdx.y;       // batch
  if (threadIdx.x == 0) {
    // searchsorted(cum, p, side='right'): first t with cum[t] > p
    int lo = 0, hi = T_;
    while (lo < hi) {
      const int mid = (lo + hi) >> 1;
      if (cum[mid] <= p) lo = mid + 1; else hi = mid;
    }
    sIdx = (lo < T_) ? lo : -1;        // lo == T_  <=>  p >= cum[-1] (invalid)
  }
  __syncthreads();
  const int idx = sIdx;

  f4 v = {0.0f, 0.0f, 0.0f, 0.0f};
  if (idx >= 0) {
    const f4* src = (const f4*)(x + ((size_t)b * T_ + idx) * D_);
    v = src[threadIdx.x];
  }
  f4* dst = (f4*)(out + ((size_t)b * TOTAL_LEN + p) * D_);
  __builtin_nontemporal_store(v, dst + threadIdx.x);
}

// ---------------------------------------------------------------------------
extern "C" void kernel_launch(void* const* d_in, const int* in_sizes, int n_in,
                              void* d_out, int out_size, void* d_ws, size_t ws_size,
                              hipStream_t stream) {
  const float* x  = (const float*)d_in[0];   // (B,T,D) f32
  const float* w1 = (const float*)d_in[1];   // (D,H)   f32
  const float* b1 = (const float*)d_in[2];   // (H,)    f32
  const float* w2 = (const float*)d_in[3];   // (H,1)   f32
  const float* b2 = (const float*)d_in[4];   // (1,)    f32
  // d_in[5] = total_length (static: B*T*8 = TOTAL_LEN)

  float* out    = (float*)d_out;                             // expanded first
  float* durOut = out + (size_t)B_ * TOTAL_LEN * D_;         // then (B,T) durations

  int* duri = (int*)d_ws;        // T_ ints: batch-0 integer durations
  int* cum  = duri + T_;         // T_ ints: inclusive cumsum

  // 1) fused MLP duration predictor via WMMA (512 waves)
  dur_mlp_wmma<<<(B_ * T_) / 16, 32, 0, stream>>>(x, w1, b1, w2, b2, durOut, duri);

  // 2) cumsum of batch-0 durations
  dur_scan<<<1, 32, 0, stream>>>(duri, cum);

  // 3) expand / length-regulate (bandwidth-bound part)
  dim3 grid(TOTAL_LEN, B_);
  expand_k<<<grid, 96, 0, stream>>>(x, cum, out);
}
